// GumbelSoftmax_6657199309125
// MI455X (gfx1250) — matrix operand order repaired and checked by reading, not compiled, not run on hardware
//
#include <hip/hip_runtime.h>
#include <hip/hip_bf16.h>

// Problem constants (from reference setup_inputs)
#define BB 32
#define NN 4096
#define FF 256
#define CC 64
#define INV_TEMP 2.0f          // 1 / 0.5

typedef __attribute__((ext_vector_type(16))) __bf16 v16bf;
typedef __attribute__((ext_vector_type(8)))  __bf16 v8bf;
typedef __attribute__((ext_vector_type(8)))  float  v8f;

// ---------------------------------------------------------------------------
// Kernel 0: zero the output (harness poisons d_out with 0xAA)
// ---------------------------------------------------------------------------
__global__ void gs_zero_kernel(float* __restrict__ out, int n) {
    int i = blockIdx.x * blockDim.x + threadIdx.x;
    if (i < n) out[i] = 0.0f;
}

// ---------------------------------------------------------------------------
// Kernel 1: gumbel-softmax expected-cluster index, one wave32 per token.
// Lane l handles clusters 2l and 2l+1 (float2 loads -> 256B coalesced/wave).
// idx = trunc( sum_c y_c * c ) with y = softmax((logits+g)/T).
// ---------------------------------------------------------------------------
__global__ void gs_index_kernel(const float* __restrict__ logits,
                                const float* __restrict__ gumbel,
                                unsigned char* __restrict__ idx8) {
    const int wave = (blockIdx.x * blockDim.x + threadIdx.x) >> 5;   // token id
    const int lane = threadIdx.x & 31;

    const size_t base = (size_t)wave * CC + (size_t)(lane * 2);
    const float2 l2 = *(const float2*)(logits + base);
    const float2 g2 = *(const float2*)(gumbel + base);

    const float z0 = (l2.x + g2.x) * INV_TEMP;
    const float z1 = (l2.y + g2.y) * INV_TEMP;

    // wave max
    float m = fmaxf(z0, z1);
    #pragma unroll
    for (int off = 16; off > 0; off >>= 1)
        m = fmaxf(m, __shfl_xor(m, off, 32));

    const float e0 = __expf(z0 - m);
    const float e1 = __expf(z1 - m);
    float s = e0 + e1;                                   // sum of exp
    float w = e0 * (float)(lane * 2) + e1 * (float)(lane * 2 + 1);  // sum exp*c
    #pragma unroll
    for (int off = 16; off > 0; off >>= 1) {
        s += __shfl_xor(s, off, 32);
        w += __shfl_xor(w, off, 32);
    }

    if (lane == 0) {
        int c = (int)(w / s);                            // truncation, as astype(int32)
        c = (c < 0) ? 0 : ((c > CC - 1) ? (CC - 1) : c);
        idx8[wave] = (unsigned char)c;
    }
}

// ---------------------------------------------------------------------------
// Kernel 2: one-hot GEMM segment-sum via v_wmma_f32_16x16x32_bf16.
// Block = (batch b, F-slice of 64, K-part of 1024). 256 threads = 8 waves.
// Wave w: C-tile mt = w/2, two F-tiles nt0 = (w&1)*2, nt0+1.
// values split into bf16 hi+lo; both accumulated into the same fp32 acc.
// ---------------------------------------------------------------------------
#define KPAD 40     // halfs per LDS column: 80B stride -> 16B aligned, bank-spread

__global__ __launch_bounds__(256) void gs_segsum_wmma_kernel(
        const float* __restrict__ values,
        const unsigned char* __restrict__ idx8,
        float* __restrict__ out) {
    __shared__ __bf16 sHi[64][KPAD];          // [f_local][k] transposed tile (hi part)
    __shared__ __bf16 sLo[64][KPAD];          // (lo part)
    __shared__ unsigned char sIdx[32];

    const int tid  = threadIdx.x;
    const int lane = tid & 31;
    const int wave = tid >> 5;

    const int b      = blockIdx.x >> 4;        // 0..31
    const int kpart  = (blockIdx.x >> 2) & 3;  // 0..3  (1024 rows each)
    const int fslice = blockIdx.x & 3;         // 0..3  (64 cols each)
    const int fbase  = fslice * 64;

    const int mt  = wave >> 1;                 // C-tile 0..3
    const int nt0 = (wave & 1) * 2;            // F-tiles nt0, nt0+1 (within slice)

    v8f acc0 = {0.f, 0.f, 0.f, 0.f, 0.f, 0.f, 0.f, 0.f};
    v8f acc1 = {0.f, 0.f, 0.f, 0.f, 0.f, 0.f, 0.f, 0.f};

    const size_t vbase  = ((size_t)b * NN + (size_t)kpart * 1024) * FF + fbase;
    const int    ibase  = b * NN + kpart * 1024;

    // staging coordinates: 2048 floats/chunk, 256 thr * 2 float4
    const int sk0 = tid >> 4;                  // row 0..15
    const int sf  = (tid & 15) * 4;            // col 0..60

    // A-fragment lane geometry (16-bit A 16x32 layout)
    const int am    = lane & 15;               // M within tile
    const int kOffA = (lane & 16) ? 16 : 0;    // K base for halves 0..15
    const int cme   = mt * 16 + am;            // global cluster this lane matches
    // B-fragment lane geometry (K x N, 32x16)
    const int kOffB = (lane & 16) ? 16 : 0;
    const int bn    = lane & 15;

    const __bf16 kOne  = (__bf16)1.0f;
    const __bf16 kZero = (__bf16)0.0f;

    for (int chunk = 0; chunk < 32; ++chunk) {
        const float* src = values + vbase + (size_t)chunk * 32 * FF;

        if (tid < 32) sIdx[tid] = idx8[ibase + chunk * 32 + tid];

        // prefetch next chunk while we work on this one
        if (chunk + 1 < 32)
            __builtin_prefetch(src + 32 * FF + (size_t)sk0 * FF + sf, 0, 1);

        // stage values -> bf16 hi/lo, transposed to [f][k]
        #pragma unroll
        for (int i = 0; i < 2; ++i) {
            const int k = sk0 + i * 16;
            const float4 v4 = *(const float4*)(src + (size_t)k * FF + sf);
            const float vv[4] = {v4.x, v4.y, v4.z, v4.w};
            #pragma unroll
            for (int j = 0; j < 4; ++j) {
                const __bf16 h = (__bf16)vv[j];
                const __bf16 l = (__bf16)(vv[j] - (float)h);
                sHi[sf + j][k] = h;
                sLo[sf + j][k] = l;
            }
        }
        __syncthreads();

        // Build one-hot A fragment (exact in bf16), shared by both output tiles
        v16bf afrag;
        #pragma unroll
        for (int j = 0; j < 8; ++j) {
            afrag[j]     = ((int)sIdx[kOffA + j]      == cme) ? kOne : kZero;
            afrag[j + 8] = ((int)sIdx[8 + kOffA + j]  == cme) ? kOne : kZero;
        }

        // Tile 0
        {
            const int col = nt0 * 16 + bn;
            const v8bf* ph = (const v8bf*)&sHi[col][kOffB];
            const v8bf* pl = (const v8bf*)&sLo[col][kOffB];
            const v16bf bhi = __builtin_shufflevector(ph[0], ph[1],
                0, 1, 2, 3, 4, 5, 6, 7, 8, 9, 10, 11, 12, 13, 14, 15);
            const v16bf blo = __builtin_shufflevector(pl[0], pl[1],
                0, 1, 2, 3, 4, 5, 6, 7, 8, 9, 10, 11, 12, 13, 14, 15);
            acc0 = __builtin_amdgcn_wmma_f32_16x16x32_bf16(
                false, afrag, false, bhi, (short)0, acc0, false, false);
            acc0 = __builtin_amdgcn_wmma_f32_16x16x32_bf16(
                false, afrag, false, blo, (short)0, acc0, false, false);
        }
        // Tile 1
        {
            const int col = (nt0 + 1) * 16 + bn;
            const v8bf* ph = (const v8bf*)&sHi[col][kOffB];
            const v8bf* pl = (const v8bf*)&sLo[col][kOffB];
            const v16bf bhi = __builtin_shufflevector(ph[0], ph[1],
                0, 1, 2, 3, 4, 5, 6, 7, 8, 9, 10, 11, 12, 13, 14, 15);
            const v16bf blo = __builtin_shufflevector(pl[0], pl[1],
                0, 1, 2, 3, 4, 5, 6, 7, 8, 9, 10, 11, 12, 13, 14, 15);
            acc1 = __builtin_amdgcn_wmma_f32_16x16x32_bf16(
                false, afrag, false, bhi, (short)0, acc1, false, false);
            acc1 = __builtin_amdgcn_wmma_f32_16x16x32_bf16(
                false, afrag, false, blo, (short)0, acc1, false, false);
        }
        __syncthreads();
    }

    // Epilogue: C/D layout -> VGPR j: lanes 0-15 M=j, lanes 16-31 M=j+8, N=lane&15.
    // K-parts combine with fp32 global atomics (out pre-zeroed).
    const int mAdd = (lane & 16) ? 8 : 0;
    const int fcol = lane & 15;
    float* obase = out + ((size_t)b * CC + (size_t)(mt * 16 + mAdd)) * FF + fbase;
    #pragma unroll
    for (int j = 0; j < 8; ++j) {
        atomicAdd(obase + (size_t)j * FF + nt0 * 16 + fcol,       acc0[j]);
        atomicAdd(obase + (size_t)j * FF + (nt0 + 1) * 16 + fcol, acc1[j]);
    }
}

// ---------------------------------------------------------------------------
extern "C" void kernel_launch(void* const* d_in, const int* in_sizes, int n_in,
                              void* d_out, int out_size, void* d_ws, size_t ws_size,
                              hipStream_t stream) {
    (void)in_sizes; (void)n_in; (void)out_size; (void)ws_size;

    const float* values = (const float*)d_in[0];   // [B,N,F]
    const float* logits = (const float*)d_in[1];   // [B,N,C]
    const float* gumbel = (const float*)d_in[2];   // [B,N,C]
    float* out = (float*)d_out;                    // [B,C,F]
    unsigned char* idx8 = (unsigned char*)d_ws;    // [B*N] cluster ids

    // 1) zero output
    const int nOut = BB * CC * FF;
    gs_zero_kernel<<<(nOut + 255) / 256, 256, 0, stream>>>(out, nOut);

    // 2) indices: one wave per token -> B*N*32 threads
    const int nTok = BB * NN;
    gs_index_kernel<<<(nTok * 32) / 256, 256, 0, stream>>>(logits, gumbel, idx8);

    // 3) one-hot WMMA segment-sum: 32 batches x 4 F-slices x 4 K-parts
    gs_segsum_wmma_kernel<<<BB * 4 * 4, 256, 0, stream>>>(values, idx8, out);
}
